// SKIPGAT_22668837388774
// MI455X (gfx1250) — matrix operand order repaired and checked by reading, not compiled, not run on hardware
//
#include <hip/hip_runtime.h>
#include <hip/hip_bf16.h>
#include <stdint.h>

// ---------------- problem constants (match reference) ----------------
#define NNODES 50000        // divisible by 16 -> 3125 row tiles
#define NEDGES 800000
#define F_IN   128
#define HEADS  8
#define C1     32           // per-head channels layer 1
#define CH1    (HEADS*C1)   // 256
#define C2     16           // per-head channels layer 2
#define CH2    (HEADS*C2)   // 128
#define SLOPE  0.2f

typedef __attribute__((ext_vector_type(16))) __bf16        v16bf;
typedef __attribute__((ext_vector_type(8)))  float         v8f;
typedef __attribute__((ext_vector_type(8)))  unsigned int  v8u;

// ---------------- small device helpers ----------------
static __device__ __forceinline__ uint16_t f2bf(float f) {
  unsigned u = __float_as_uint(f);
  unsigned r = u + 0x7FFFu + ((u >> 16) & 1u);   // round-to-nearest-even
  return (uint16_t)(r >> 16);
}
static __device__ __forceinline__ float leaky(float e) { return e > 0.f ? e : SLOPE * e; }
// order-preserving float->uint key for atomicMax on signed floats
static __device__ __forceinline__ unsigned fkey(float f) {
  unsigned u = __float_as_uint(f);
  return (u & 0x80000000u) ? ~u : (u | 0x80000000u);
}
static __device__ __forceinline__ float funkey(unsigned k) {
  unsigned u = (k & 0x80000000u) ? (k & 0x7FFFFFFFu) : ~k;
  return __uint_as_float(u);
}

// ---------------- prep: relu + cast to bf16 ----------------
__global__ void k_relu_bf16(const float* __restrict__ in, uint16_t* __restrict__ out, int n) {
  int t = blockIdx.x * blockDim.x + threadIdx.x;
  if (t < n) { float v = in[t]; out[t] = f2bf(v > 0.f ? v : 0.f); }
}

// transpose W [K,Nout] f32 -> Wt [Nout,K] bf16 (makes B K-pairs contiguous)
__global__ void k_wt_bf16(const float* __restrict__ W, uint16_t* __restrict__ Wt, int K, int Nout) {
  int t = blockIdx.x * blockDim.x + threadIdx.x;
  if (t < K * Nout) {
    int k = t / Nout, n = t - k * Nout;
    Wt[(size_t)n * K + k] = f2bf(W[t]);
  }
}

// ---------------- bf16 WMMA GEMM: C[M,Nout] = A[M,K] * Bt[Nout,K]^T ----------------
// Register-blocked: each wave computes a 16 x (16*NB) strip. The A fragment is
// loaded once per K-step and reused across NB back-to-back WMMAs (no D->A/B
// hazard between them, so they pipeline).
// A fragment layout (16-bit A 16x32, wave32): lane = half*16 + m;
//   VGPR v(0..3) = K pair (2v + half*8); v(4..7) = pair (16 + 2(v-4) + half*8).
// B fragment: lane = half*16 + n; VGPR v: same K pairs as A, of column n.
template <int K, int NB>
__global__ void k_gemm_wmma(const uint16_t* __restrict__ A, const uint16_t* __restrict__ Bt,
                            float* __restrict__ C, int M, int Nout) {
  const int lane = threadIdx.x & 31;
  const int wave = threadIdx.x >> 5;
  const int half = lane >> 4;
  const int l16  = lane & 15;
  const int m0 = (blockIdx.y * (blockDim.x >> 5) + wave) * 16;
  if (m0 >= M) return;
  const int n0 = blockIdx.x * (16 * NB);

  const uint16_t* arow = A + (size_t)(m0 + l16) * K;
  const uint16_t* brow[NB];
#pragma unroll
  for (int j = 0; j < NB; ++j)
    brow[j] = Bt + (size_t)(n0 + j * 16 + l16) * K;

  v8f acc[NB];
#pragma unroll
  for (int j = 0; j < NB; ++j)
#pragma unroll
    for (int i = 0; i < 8; ++i) acc[j][i] = 0.f;

#pragma unroll
  for (int k0 = 0; k0 < K; k0 += 32) {
    v8u au;
#pragma unroll
    for (int v = 0; v < 4; ++v) {
      au[v]     = *(const unsigned*)(arow + k0 + half * 8 + 2 * v);
      au[v + 4] = *(const unsigned*)(arow + k0 + 16 + half * 8 + 2 * v);
    }
    v16bf a = __builtin_bit_cast(v16bf, au);
#pragma unroll
    for (int j = 0; j < NB; ++j) {
      v8u bu;
#pragma unroll
      for (int v = 0; v < 8; ++v)
        bu[v] = *(const unsigned*)(brow[j] + k0 + half * 8 + 2 * v + (v >= 4 ? 8 : 0));
      v16bf b = __builtin_bit_cast(v16bf, bu);
      acc[j] = __builtin_amdgcn_wmma_f32_16x16x32_bf16(false, a, false, b, (short)0,
                                                       acc[j], false, false);
    }
  }
  // C/D layout: lane -> col + l16; VGPR v -> row m0 + v + half*8
#pragma unroll
  for (int j = 0; j < NB; ++j) {
    float* crow = C + (size_t)m0 * Nout + n0 + j * 16 + l16;
#pragma unroll
    for (int v = 0; v < 8; ++v)
      crow[(size_t)(v + half * 8) * Nout] = acc[j][v];
  }
}

// ---------------- attention logits per node (wave32 dot + shfl reduce) ----------------
template <int C>
__global__ void k_attn_logits(const float* __restrict__ H, const float* __restrict__ asrc,
                              const float* __restrict__ adst, float* __restrict__ als,
                              float* __restrict__ ald, unsigned* __restrict__ mk, int n) {
  const int lane = threadIdx.x & 31;
  const int node = blockIdx.x * (blockDim.x >> 5) + (threadIdx.x >> 5);
  if (node >= n) return;
  const int CH = C * HEADS;
#pragma unroll
  for (int h = 0; h < HEADS; ++h) {
    float vs = 0.f, vd = 0.f;
#pragma unroll
    for (int c = lane; c < C; c += 32) {
      float hv = H[(size_t)node * CH + h * C + c];
      vs += hv * asrc[h * C + c];
      vd += hv * adst[h * C + c];
    }
#pragma unroll
    for (int off = 16; off; off >>= 1) { vs += __shfl_xor(vs, off); vd += __shfl_xor(vd, off); }
    if (lane == 0) {
      als[node * HEADS + h] = vs;
      ald[node * HEADS + h] = vd;
      // self-loop logit seeds the running max
      mk[node * HEADS + h] = fkey(leaky(vs + vd));
    }
  }
}

// ---------------- softmax pass 1: per-edge atomic max ----------------
__global__ void k_edge_max(const int* __restrict__ src, const int* __restrict__ dst,
                           const float* __restrict__ als, const float* __restrict__ ald,
                           unsigned* __restrict__ mk) {
  int t = blockIdx.x * blockDim.x + threadIdx.x;
  if (t >= NEDGES * HEADS) return;
  int e = t >> 3, h = t & 7;
  int s = src[e], d = dst[e];
  float lg = leaky(als[s * HEADS + h] + ald[d * HEADS + h]);
  atomicMax(&mk[d * HEADS + h], fkey(lg));
}

// decode max key in place (-> float) and seed denominator with the self-loop term
__global__ void k_node_max(unsigned* __restrict__ mk, const float* __restrict__ als,
                           const float* __restrict__ ald, float* __restrict__ denom, int n) {
  int t = blockIdx.x * blockDim.x + threadIdx.x;
  if (t >= n * HEADS) return;
  float m = funkey(mk[t]);
  ((float*)mk)[t] = m;
  denom[t] = __expf(leaky(als[t] + ald[t]) - m);
}

// ---------------- softmax pass 2: exp + atomic denominator ----------------
__global__ void k_edge_exp(const int* __restrict__ src, const int* __restrict__ dst,
                           const float* __restrict__ als, const float* __restrict__ ald,
                           const float* __restrict__ mfl, float* __restrict__ denom,
                           float* __restrict__ ex) {
  int t = blockIdx.x * blockDim.x + threadIdx.x;
  if (t >= NEDGES * HEADS) return;
  int e = t >> 3, h = t & 7;
  int s = src[e], d = dst[e];
  float lg = leaky(als[s * HEADS + h] + ald[d * HEADS + h]);
  float v = __expf(lg - mfl[d * HEADS + h]);
  ex[t] = v;
  atomicAdd(&denom[d * HEADS + h], v);
}

// ---------------- init out = bias + self-loop message ----------------
template <int CH>
__global__ void k_node_init(const float* __restrict__ H, const float* __restrict__ bias,
                            const float* __restrict__ als, const float* __restrict__ ald,
                            const float* __restrict__ mfl, const float* __restrict__ denom,
                            float* __restrict__ O, int n) {
  int t = blockIdx.x * blockDim.x + threadIdx.x;
  if (t >= n * CH) return;
  int node = t / CH;
  int c = t - node * CH;
  int h = c / (CH / HEADS);
  int nh = node * HEADS + h;
  float w = __expf(leaky(als[nh] + ald[nh]) - mfl[nh]) / denom[nh];
  O[t] = bias[c] + H[t] * w;
}

// ---------------- pass 3: gather h[src], scale by alpha, atomic scatter ----------------
template <int CH>
__global__ void k_edge_agg(const int* __restrict__ src, const int* __restrict__ dst,
                           const float* __restrict__ ex, const float* __restrict__ denom,
                           const float* __restrict__ H, float* __restrict__ O) {
  const int lane = threadIdx.x & 31;
  const int e = blockIdx.x * (blockDim.x >> 5) + (threadIdx.x >> 5);
  if (e >= NEDGES) return;
  int s = src[e], d = dst[e];
  const float* hs = H + (size_t)s * CH;
  float* od = O + (size_t)d * CH;
#pragma unroll
  for (int c = lane; c < CH; c += 32) {
    int h = c / (CH / HEADS);
    float alpha = ex[e * HEADS + h] / denom[d * HEADS + h];
    atomicAdd(&od[c], hs[c] * alpha);
  }
}

// ---------------- residual add ----------------
__global__ void k_residual(const float* __restrict__ x, float* __restrict__ out, int n) {
  int t = blockIdx.x * blockDim.x + threadIdx.x;
  if (t < n) out[t] += x[t];
}

static inline int gridOf(long long n, int b) { return (int)((n + b - 1) / b); }

extern "C" void kernel_launch(void* const* d_in, const int* in_sizes, int n_in,
                              void* d_out, int out_size, void* d_ws, size_t ws_size,
                              hipStream_t stream) {
  (void)in_sizes; (void)n_in; (void)out_size; (void)ws_size;
  const float* x   = (const float*)d_in[0];
  const int*   eix = (const int*)d_in[1];
  /* d_in[2] edge_weight is ignored by the reference (edge_dim=None) */
  const float* W1  = (const float*)d_in[3];
  const float* a1s = (const float*)d_in[4];
  const float* a1d = (const float*)d_in[5];
  const float* b1  = (const float*)d_in[6];
  const float* W2  = (const float*)d_in[7];
  const float* a2s = (const float*)d_in[8];
  const float* a2d = (const float*)d_in[9];
  const float* b2  = (const float*)d_in[10];
  const int* src = eix;
  const int* dst = eix + NEDGES;

  // ---- carve workspace (~200 MB) ----
  char* ws = (char*)d_ws;
  size_t off = 0;
  auto take = [&](size_t bytes) -> void* {
    void* p = ws + off;
    off += (bytes + 255) & ~(size_t)255;
    return p;
  };
  uint16_t* h0b = (uint16_t*)take((size_t)NNODES * F_IN * 2);   // bf16 relu(x)
  uint16_t* wt1 = (uint16_t*)take((size_t)CH1 * F_IN * 2);      // W1^T bf16
  uint16_t* wt2 = (uint16_t*)take((size_t)CH2 * CH1 * 2);       // W2^T bf16
  float*    H1  = (float*)take((size_t)NNODES * CH1 * 4);       // x' @ W1
  float*    O1  = (float*)take((size_t)NNODES * CH1 * 4);       // layer-1 output
  uint16_t* h1b = (uint16_t*)take((size_t)NNODES * CH1 * 2);    // bf16 relu(O1)
  float*    H2  = (float*)take((size_t)NNODES * CH2 * 4);       // h1 @ W2
  float*    al1s = (float*)take((size_t)NNODES * HEADS * 4);
  float*    al1d = (float*)take((size_t)NNODES * HEADS * 4);
  unsigned* MK1  = (unsigned*)take((size_t)NNODES * HEADS * 4); // max key -> max float
  float*    D1   = (float*)take((size_t)NNODES * HEADS * 4);
  float*    al2s = (float*)take((size_t)NNODES * HEADS * 4);
  float*    al2d = (float*)take((size_t)NNODES * HEADS * 4);
  unsigned* MK2  = (unsigned*)take((size_t)NNODES * HEADS * 4);
  float*    D2   = (float*)take((size_t)NNODES * HEADS * 4);
  float*    EX   = (float*)take((size_t)NEDGES * HEADS * 4);    // reused by both layers

  float* out = (float*)d_out;  // [N, 128]; used as layer-2 accumulator

  const int B = 256;
  // ---- prep ----
  k_relu_bf16<<<gridOf((long long)NNODES * F_IN, B), B, 0, stream>>>(x, h0b, NNODES * F_IN);
  k_wt_bf16<<<gridOf((long long)F_IN * CH1, B), B, 0, stream>>>(W1, wt1, F_IN, CH1);
  k_wt_bf16<<<gridOf((long long)CH1 * CH2, B), B, 0, stream>>>(W2, wt2, CH1, CH2);

  // ---- layer 1 ----
  // 4 waves/block (4 m-tiles), each wave covers a 16x64 strip (NB=4).
  k_gemm_wmma<F_IN, 4><<<dim3(CH1 / 64, gridOf(NNODES / 16, 4)), 128, 0, stream>>>(
      h0b, wt1, H1, NNODES, CH1);
  k_attn_logits<C1><<<gridOf(NNODES, 8), 256, 0, stream>>>(H1, a1s, a1d, al1s, al1d, MK1, NNODES);
  k_edge_max<<<gridOf((long long)NEDGES * HEADS, B), B, 0, stream>>>(src, dst, al1s, al1d, MK1);
  k_node_max<<<gridOf((long long)NNODES * HEADS, B), B, 0, stream>>>(MK1, al1s, al1d, D1, NNODES);
  k_edge_exp<<<gridOf((long long)NEDGES * HEADS, B), B, 0, stream>>>(src, dst, al1s, al1d,
                                                                    (float*)MK1, D1, EX);
  k_node_init<CH1><<<gridOf((long long)NNODES * CH1, B), B, 0, stream>>>(H1, b1, al1s, al1d,
                                                                         (float*)MK1, D1, O1, NNODES);
  k_edge_agg<CH1><<<gridOf(NEDGES, 8), 256, 0, stream>>>(src, dst, EX, D1, H1, O1);

  // ---- layer 2 ----
  k_relu_bf16<<<gridOf((long long)NNODES * CH1, B), B, 0, stream>>>(O1, h1b, NNODES * CH1);
  k_gemm_wmma<CH1, 4><<<dim3(CH2 / 64, gridOf(NNODES / 16, 4)), 128, 0, stream>>>(
      h1b, wt2, H2, NNODES, CH2);
  k_attn_logits<C2><<<gridOf(NNODES, 8), 256, 0, stream>>>(H2, a2s, a2d, al2s, al2d, MK2, NNODES);
  k_edge_max<<<gridOf((long long)NEDGES * HEADS, B), B, 0, stream>>>(src, dst, al2s, al2d, MK2);
  k_node_max<<<gridOf((long long)NNODES * HEADS, B), B, 0, stream>>>(MK2, al2s, al2d, D2, NNODES);
  k_edge_exp<<<gridOf((long long)NEDGES * HEADS, B), B, 0, stream>>>(src, dst, al2s, al2d,
                                                                    (float*)MK2, D2, EX);
  k_node_init<CH2><<<gridOf((long long)NNODES * CH2, B), B, 0, stream>>>(H2, b2, al2s, al2d,
                                                                         (float*)MK2, D2, out, NNODES);
  k_edge_agg<CH2><<<gridOf(NEDGES, 8), 256, 0, stream>>>(src, dst, EX, D2, H2, out);

  // ---- residual ----
  k_residual<<<gridOf((long long)NNODES * CH2, B), B, 0, stream>>>(x, out, NNODES * CH2);
}
